// CausalMultiHeadSelfAttention_62929860821099
// MI455X (gfx1250) — matrix operand order repaired and checked by reading, not compiled, not run on hardware
//
#include <hip/hip_runtime.h>

#define DIMSZ 1024
#define NHEAD 16
#define HDIM  64
#define BATCH 2
#define SEQ   2048
#define MROWS (BATCH * SEQ) /* 4096 */

typedef __attribute__((ext_vector_type(16))) __bf16 v16bf;
typedef __attribute__((ext_vector_type(8)))  __bf16 v8bf;
typedef __attribute__((ext_vector_type(4)))  __bf16 v4bf;
typedef __attribute__((ext_vector_type(8)))  float  v8f;
typedef __attribute__((ext_vector_type(4)))  unsigned v4u;
typedef __attribute__((ext_vector_type(8)))  int    v8i;
typedef __attribute__((ext_vector_type(4)))  int    v4i;

// ---------- helpers ----------

static __device__ __forceinline__ __bf16 f2bf(float f) { return (__bf16)f; }

static __device__ __forceinline__ v16bf combine(v8bf lo, v8bf hi) {
  v16bf r;
#pragma unroll
  for (int i = 0; i < 8; ++i) { r[i] = lo[i]; r[8 + i] = hi[i]; }
  return r;
}

static __device__ __forceinline__ v8bf ld8(const __bf16* p) { return *(const v8bf*)p; }

static __device__ __forceinline__ float redmax16(float v) {
#pragma unroll
  for (int m = 1; m < 16; m <<= 1) v = fmaxf(v, __shfl_xor(v, m, 32));
  return v;
}
static __device__ __forceinline__ float redsum16(float v) {
#pragma unroll
  for (int m = 1; m < 16; m <<= 1) v += __shfl_xor(v, m, 32);
  return v;
}

// LDS byte offset (workgroup-relative) for TDM descriptors
static __device__ __forceinline__ unsigned lds_offset(const void* p) {
  return (unsigned)(unsigned long long)(__attribute__((address_space(3))) const char*)p;
}

// TDM: async 2D tile load Global->LDS, bf16 elements. Dims/strides in elements.
static __device__ __forceinline__ void tdm_load_2d(unsigned lds_off, const void* gptr,
                                                   unsigned tensor_d0, unsigned tensor_d1,
                                                   unsigned tile_d0, unsigned tile_d1,
                                                   unsigned stride0) {
  unsigned long long ga = (unsigned long long)(size_t)gptr;
  v4u g0;
  g0[0] = 1u;                                             // count=1 (valid), user mode
  g0[1] = lds_off;                                        // lds_addr (bytes)
  g0[2] = (unsigned)(ga & 0xffffffffu);                   // global_addr[31:0]
  g0[3] = (unsigned)((ga >> 32) & 0x01ffffffu) | (2u << 30); // addr[56:32] | type=2
  v8i g1;
  g1[0] = (int)(1u << 16);                                // wg_mask=0, data_size=1 (2B)
  g1[1] = (int)((tensor_d0 & 0xffffu) << 16);             // tensor_dim0[15:0] @ bit48
  g1[2] = (int)(((tensor_d0 >> 16) & 0xffffu) | ((tensor_d1 & 0xffffu) << 16));
  g1[3] = (int)(((tensor_d1 >> 16) & 0xffffu) | ((tile_d0 & 0xffffu) << 16));
  g1[4] = (int)(tile_d1 & 0xffffu);                       // tile_dim1, tile_dim2=0
  g1[5] = (int)stride0;                                   // tensor_dim0_stride[31:0]
  g1[6] = 0;                                              // stride0[47:32] | dim1_stride lo
  g1[7] = 0;
  v4i z4 = {0, 0, 0, 0};
#if defined(__clang_major__) && (__clang_major__ >= 23)
  v8i z8 = {};
  __builtin_amdgcn_tensor_load_to_lds(g0, g1, z4, z4, z8, 0);
#else
  __builtin_amdgcn_tensor_load_to_lds(g0, g1, z4, z4, 0);
#endif
}

// ---------- kernel 0: one-shot f32 -> bf16 conversion (bandwidth-bound) ----------

__global__ __launch_bounds__(256) void cvt_kernel(const float* __restrict__ src,
                                                  __bf16* __restrict__ dst, int n4) {
  int i = blockIdx.x * blockDim.x + threadIdx.x;
  if (i < n4) {
    float4 v = *(const float4*)(src + (size_t)i * 4);
    v4bf o;
    o[0] = f2bf(v.x); o[1] = f2bf(v.y); o[2] = f2bf(v.z); o[3] = f2bf(v.w);
    *(v4bf*)(dst + (size_t)i * 4) = o;
  }
}

// ---------- kernel 1: qkv = x @ w_qkv.T  (bf16 WMMA), scatter to Q/K/Vt ----------
// block = 256 threads (8 waves); wave computes 32 rows x 64 cols (B frags reused 2x).

__global__ __launch_bounds__(256) void qkv_kernel(const __bf16* __restrict__ x,
                                                  const __bf16* __restrict__ w,
                                                  __bf16* __restrict__ qws,
                                                  __bf16* __restrict__ kws,
                                                  __bf16* __restrict__ vtws) {
  const int lane  = threadIdx.x & 31;
  const int wave  = threadIdx.x >> 5;
  const int lhalf = lane >> 4;
  const int l16   = lane & 15;
  const int r0    = blockIdx.x * 32;
  const int c0    = blockIdx.y * 512 + wave * 64;

  v8f acc[2][4] = {};
  const __bf16* xrow0 = x + (size_t)(r0 + l16) * DIMSZ;
  const __bf16* xrow1 = xrow0 + (size_t)16 * DIMSZ;
  const int aoff = lhalf ? 8 : 0;
  const int boff = lhalf ? 8 : 0;

  for (int k0 = 0; k0 < DIMSZ; k0 += 32) {
    v16bf a0 = combine(ld8(xrow0 + k0 + aoff), ld8(xrow0 + k0 + aoff + 16));
    v16bf a1 = combine(ld8(xrow1 + k0 + aoff), ld8(xrow1 + k0 + aoff + 16));
#pragma unroll
    for (int nt = 0; nt < 4; ++nt) {
      const __bf16* wrow = w + (size_t)(c0 + nt * 16 + l16) * DIMSZ + k0 + boff;
      v16bf bf = combine(ld8(wrow), ld8(wrow + 16));
      acc[0][nt] = __builtin_amdgcn_wmma_f32_16x16x32_bf16(false, a0, false, bf,
                                                           (short)0, acc[0][nt], false, false);
      acc[1][nt] = __builtin_amdgcn_wmma_f32_16x16x32_bf16(false, a1, false, bf,
                                                           (short)0, acc[1][nt], false, false);
    }
  }

  // scatter store: col c -> {q,k,v}[b][h][t][d] (v transposed)
#pragma unroll
  for (int mt = 0; mt < 2; ++mt)
#pragma unroll
    for (int nt = 0; nt < 4; ++nt)
#pragma unroll
      for (int r = 0; r < 8; ++r) {
        int m   = r + (lhalf ? 8 : 0);
        int row = r0 + mt * 16 + m;
        int b   = row >> 11;          // / SEQ
        int t   = row & (SEQ - 1);
        int c   = c0 + nt * 16 + l16;
        int sel = c >> 10;            // 0=q 1=k 2=v
        int cc  = c & (DIMSZ - 1);
        int h   = cc >> 6;
        int d   = cc & 63;
        int bh  = b * NHEAD + h;
        __bf16 val = f2bf(acc[mt][nt][r]);
        if (sel == 0)      qws[((size_t)bh * SEQ + t) * HDIM + d] = val;
        else if (sel == 1) kws[((size_t)bh * SEQ + t) * HDIM + d] = val;
        else               vtws[((size_t)bh * HDIM + d) * SEQ + t] = val;
      }
}

// ---------- kernel 2: flash attention, one wave per (bh, 16-query tile) ----------
// K/V blocks are double-buffered in LDS via the Tensor Data Mover (TENSORcnt).

__global__ __launch_bounds__(32) void attn_kernel(const __bf16* __restrict__ qws,
                                                  const __bf16* __restrict__ kws,
                                                  const __bf16* __restrict__ vtws,
                                                  __bf16* __restrict__ aows) {
  __shared__ __bf16 klds[2][32 * HDIM];   // 2 x 4 KB : K block, 32 keys x 64 d
  __shared__ __bf16 vlds[2][HDIM * 32];   // 2 x 4 KB : V^T block, 64 d-rows x 32 keys
  __shared__ __bf16 plds[16 * 32];        // 1 KB    : P tile

  const int lane  = threadIdx.x & 31;
  const int lhalf = lane >> 4;
  const int l16   = lane & 15;
  const int bh    = blockIdx.x;
  const int qt0   = blockIdx.y * 16;

  const __bf16* qb = qws + (size_t)bh * SEQ * HDIM;
  const __bf16* kb = kws + (size_t)bh * SEQ * HDIM;
  const __bf16* vb = vtws + (size_t)bh * HDIM * SEQ;

  const int aoff = lhalf ? 8 : 0;
  const int boff = lhalf ? 8 : 0;

  const int nblocks = (qt0 + 16 + 31) >> 5;

  // prefetch block 0 K/V tiles into LDS buffer 0 via TDM
  tdm_load_2d(lds_offset(&klds[0][0]), kb, HDIM, 32, HDIM, 32, HDIM);
  tdm_load_2d(lds_offset(&vlds[0][0]), vb, 32, HDIM, 32, HDIM, SEQ);

  // Q fragments (A-layout), resident for the whole pass
  v16bf aq[2];
  {
    const __bf16* qrow = qb + (size_t)(qt0 + l16) * HDIM;
#pragma unroll
    for (int dc = 0; dc < 2; ++dc) {
      int base = dc * 32 + aoff;
      aq[dc] = combine(ld8(qrow + base), ld8(qrow + base + 16));
    }
  }

  float rowM[8], rowL[8];
#pragma unroll
  for (int r = 0; r < 8; ++r) { rowM[r] = -1e30f; rowL[r] = 0.0f; }
  v8f o[4] = {};
  const float scale = 0.125f;            // 1/sqrt(64)

  for (int blk = 0; blk < nblocks; ++blk) {
    const int kb0 = blk * 32;
    const int cur = blk & 1;

    // prefetch next block into the other buffer, then wait for current pair
    if (blk + 1 < nblocks) {
      const int nb0 = kb0 + 32;
      tdm_load_2d(lds_offset(&klds[cur ^ 1][0]), kb + (size_t)nb0 * HDIM,
                  HDIM, 32, HDIM, 32, HDIM);
      tdm_load_2d(lds_offset(&vlds[cur ^ 1][0]), vb + nb0,
                  32, HDIM, 32, HDIM, SEQ);
      __builtin_amdgcn_s_wait_tensorcnt(2);
    } else {
      __builtin_amdgcn_s_wait_tensorcnt(0);
    }

    // S = Q * K^T for 32 keys (frags gathered from LDS)
    v8f sfr[2] = {};
#pragma unroll
    for (int si = 0; si < 2; ++si) {
#pragma unroll
      for (int dc = 0; dc < 2; ++dc) {
        const __bf16* krow = &klds[cur][(si * 16 + l16) * HDIM + dc * 32 + boff];
        v16bf bk = combine(ld8(krow), ld8(krow + 16));
        sfr[si] = __builtin_amdgcn_wmma_f32_16x16x32_bf16(false, aq[dc], false, bk,
                                                          (short)0, sfr[si], false, false);
      }
    }

    // scale + causal mask + online softmax (row = vgpr index within lane half)
#pragma unroll
    for (int r = 0; r < 8; ++r) {
      const int m = r + (lhalf ? 8 : 0);
      const int q = qt0 + m;
#pragma unroll
      for (int si = 0; si < 2; ++si) {
        int j = kb0 + si * 16 + l16;
        float v = sfr[si][r] * scale;
        sfr[si][r] = (j > q) ? -1e30f : v;
      }
      float rmax = redmax16(fmaxf(sfr[0][r], sfr[1][r]));
      float newM = fmaxf(rowM[r], rmax);
      float alpha = __expf(rowM[r] - newM);
      rowM[r] = newM;
      float p0 = __expf(sfr[0][r] - newM);
      float p1 = __expf(sfr[1][r] - newM);
      sfr[0][r] = p0; sfr[1][r] = p1;
      rowL[r] = rowL[r] * alpha + redsum16(p0 + p1);
#pragma unroll
      for (int rf = 0; rf < 4; ++rf) o[rf][r] *= alpha;
    }

    // transpose P (C-layout -> A-layout) through LDS
#pragma unroll
    for (int si = 0; si < 2; ++si)
#pragma unroll
      for (int r = 0; r < 8; ++r) {
        int m = r + (lhalf ? 8 : 0);
        plds[m * 32 + si * 16 + l16] = f2bf(sfr[si][r]);
      }
    __syncthreads();

    const __bf16* prow = plds + l16 * 32 + aoff;
    v16bf pa = combine(ld8(prow), ld8(prow + 16));

    // O += P * V  (V^T rows contiguous in LDS)
#pragma unroll
    for (int rf = 0; rf < 4; ++rf) {
      const __bf16* vrow = &vlds[cur][(rf * 16 + l16) * 32 + boff];
      v16bf bv = combine(ld8(vrow), ld8(vrow + 16));
      o[rf] = __builtin_amdgcn_wmma_f32_16x16x32_bf16(false, pa, false, bv,
                                                      (short)0, o[rf], false, false);
    }
    __syncthreads();
  }

  // epilogue: normalize, store bf16 attention output [b*T + t][h*64 + d]
  const int b = bh >> 4;
  const int h = bh & 15;
#pragma unroll
  for (int r = 0; r < 8; ++r) {
    const int m = r + (lhalf ? 8 : 0);
    const float inv = 1.0f / rowL[r];
    const size_t rowbase = (size_t)(b * SEQ + qt0 + m) * DIMSZ + h * HDIM;
#pragma unroll
    for (int rf = 0; rf < 4; ++rf)
      aows[rowbase + rf * 16 + l16] = f2bf(o[rf][r] * inv);
  }
}

// ---------- kernel 3: out = ao @ w_out.T  (bf16 WMMA, f32 coalesced stores) ----------

__global__ __launch_bounds__(256) void out_kernel(const __bf16* __restrict__ ao,
                                                  const __bf16* __restrict__ w,
                                                  float* __restrict__ out) {
  const int lane  = threadIdx.x & 31;
  const int wave  = threadIdx.x >> 5;
  const int lhalf = lane >> 4;
  const int l16   = lane & 15;
  const int r0    = blockIdx.x * 32;
  const int c0    = blockIdx.y * 512 + wave * 64;

  v8f acc[2][4] = {};
  const __bf16* arow0 = ao + (size_t)(r0 + l16) * DIMSZ;
  const __bf16* arow1 = arow0 + (size_t)16 * DIMSZ;
  const int aoff = lhalf ? 8 : 0;
  const int boff = lhalf ? 8 : 0;

  for (int k0 = 0; k0 < DIMSZ; k0 += 32) {
    v16bf a0 = combine(ld8(arow0 + k0 + aoff), ld8(arow0 + k0 + aoff + 16));
    v16bf a1 = combine(ld8(arow1 + k0 + aoff), ld8(arow1 + k0 + aoff + 16));
#pragma unroll
    for (int nt = 0; nt < 4; ++nt) {
      const __bf16* wrow = w + (size_t)(c0 + nt * 16 + l16) * DIMSZ + k0 + boff;
      v16bf bf = combine(ld8(wrow), ld8(wrow + 16));
      acc[0][nt] = __builtin_amdgcn_wmma_f32_16x16x32_bf16(false, a0, false, bf,
                                                           (short)0, acc[0][nt], false, false);
      acc[1][nt] = __builtin_amdgcn_wmma_f32_16x16x32_bf16(false, a1, false, bf,
                                                           (short)0, acc[1][nt], false, false);
    }
  }

#pragma unroll
  for (int mt = 0; mt < 2; ++mt)
#pragma unroll
    for (int nt = 0; nt < 4; ++nt)
#pragma unroll
      for (int r = 0; r < 8; ++r) {
        int m = r + (lhalf ? 8 : 0);
        out[(size_t)(r0 + mt * 16 + m) * DIMSZ + c0 + nt * 16 + l16] = acc[mt][nt][r];
      }
}

// ---------- launch ----------

extern "C" void kernel_launch(void* const* d_in, const int* in_sizes, int n_in,
                              void* d_out, int out_size, void* d_ws, size_t ws_size,
                              hipStream_t stream) {
  (void)in_sizes; (void)n_in; (void)out_size; (void)ws_size;
  const float* x     = (const float*)d_in[0];
  const float* w_qkv = (const float*)d_in[1];
  const float* w_out = (const float*)d_in[2];
  float* out = (float*)d_out;

  const size_t xelems  = (size_t)MROWS * DIMSZ;      // 4,194,304
  const size_t wqelems = (size_t)3 * DIMSZ * DIMSZ;  // 3,145,728
  const size_t woelems = (size_t)DIMSZ * DIMSZ;      // 1,048,576

  __bf16* xbf    = (__bf16*)d_ws;          // 8 MB (reused as aows after kernel 1)
  __bf16* wqkvbf = xbf + xelems;           // 6 MB
  __bf16* woutbf = wqkvbf + wqelems;       // 2 MB
  __bf16* qws    = woutbf + woelems;       // 8 MB
  __bf16* kws    = qws + xelems;           // 8 MB
  __bf16* vtws   = kws + xelems;           // 8 MB  -> 40 MB total
  __bf16* aows   = xbf;                    // alias: x_bf16 dead after kernel 1

  cvt_kernel<<<dim3((unsigned)(xelems / 4 / 256)),  dim3(256), 0, stream>>>(x,     xbf,    (int)(xelems / 4));
  cvt_kernel<<<dim3((unsigned)(wqelems / 4 / 256)), dim3(256), 0, stream>>>(w_qkv, wqkvbf, (int)(wqelems / 4));
  cvt_kernel<<<dim3((unsigned)(woelems / 4 / 256)), dim3(256), 0, stream>>>(w_out, woutbf, (int)(woelems / 4));

  dim3 blk(256);
  dim3 g1(MROWS / 32, (3 * DIMSZ) / 512);       // (128, 6)
  qkv_kernel<<<g1, blk, 0, stream>>>(xbf, wqkvbf, qws, kws, vtws);

  dim3 g2(BATCH * NHEAD, SEQ / 16);             // (32, 128)
  attn_kernel<<<g2, dim3(32), 0, stream>>>(qws, kws, vtws, aows);

  dim3 g3(MROWS / 32, DIMSZ / 512);             // (128, 2)
  out_kernel<<<g3, blk, 0, stream>>>(aows, woutbf, out);
}
